// GatedDynamicProjection_19567871001006
// MI455X (gfx1250) — matrix operand order repaired and checked by reading, not compiled, not run on hardware
//
#include <hip/hip_runtime.h>
#include <hip/hip_bf16.h>
#include <math.h>

#define NNODES 65536
#define DSZ 256
#define DVZ 64
#define RRZ 64
#define NBATCH 8

typedef __attribute__((ext_vector_type(16))) _Float16 v16h;
typedef __attribute__((ext_vector_type(8)))  _Float16 v8h;
typedef __attribute__((ext_vector_type(8)))  float    v8f;
typedef __attribute__((ext_vector_type(4)))  unsigned int u32x4;
typedef __attribute__((ext_vector_type(8)))  int i32x8;
typedef __attribute__((ext_vector_type(4)))  int i32x4;

// ---------------- WMMA helpers ----------------
__device__ __forceinline__ v8f wmma_f16(v16h a, v16h b, v8f c) {
  return __builtin_amdgcn_wmma_f32_16x16x32_f16(false, a, false, b, (short)0, c,
                                                false, false);
}
__device__ __forceinline__ float sigf(float x) { return 1.0f / (1.0f + __expf(-x)); }

// A-fragment (16x32 f16) from row-major f16 [rows x stride].
// ISA layout: lane<16 holds row=lane, K={0..7,16..23}; lane>=16 same row,
// K={8..15,24..31}  => two contiguous 8-half (16B) loads per lane.
__device__ __forceinline__ v16h a_frag(const _Float16* src, int stride, int row,
                                       int k0, int hi) {
  const _Float16* p = src + (size_t)row * stride + k0 + 8 * hi;
  v8h lo = *(const v8h*)p;
  v8h hh = *(const v8h*)(p + 16);
  v16h r;
#pragma unroll
  for (int i = 0; i < 8; i++) { r[i] = lo[i]; r[8 + i] = hh[i]; }
  return r;
}

// B-fragment from pre-swizzled tile storage (512 halfs/tile, lane*16 contiguous)
__device__ __forceinline__ v16h b_swz(const _Float16* tile, int lane) {
  return *(const v16h*)(tile + lane * 16);
}

// B-fragment gathered from row-major f16 [K x nOut] (used on LDS-staged data)
__device__ __forceinline__ v16h b_gather(const _Float16* src, int nOut, int k0,
                                         int n0, int lane) {
  int n = n0 + (lane & 15);
  int kb = k0 + 8 * (lane >> 4);
  v16h r;
#pragma unroll
  for (int h = 0; h < 16; h++) r[h] = src[(kb + (h & 7) + 16 * (h >> 3)) * nOut + n];
  return r;
}

// ---------------- weight swizzle: W[nOut][K] fp32 -> B-frag tiles f16 -------------
__global__ void k_swz(const float* __restrict__ W, _Float16* __restrict__ dst,
                      int K, int Nout) {
  int idx = blockIdx.x * blockDim.x + threadIdx.x;
  if (idx >= K * Nout) return;
  int tile = idx >> 9, rem = idx & 511;
  int lane = rem >> 4, h = rem & 15;
  int ntiles = Nout >> 4;
  int kc = tile / ntiles, nc = tile % ntiles;
  int n = nc * 16 + (lane & 15);
  int k = kc * 32 + (h & 7) + 16 * (h >> 3) + 8 * (lane >> 4);
  dst[idx] = (_Float16)W[(size_t)n * K + k];
}

// ---------------- input conversion: x_s -> f16 row-major; x_v -> [n*3+c][d] f16 ---
__global__ void k_prep_x(const float* __restrict__ xs, const float* __restrict__ xv,
                         _Float16* __restrict__ xs16, _Float16* __restrict__ xvt16) {
  size_t i = blockIdx.x * (size_t)blockDim.x + threadIdx.x;
  if (i < (size_t)NNODES * DSZ) xs16[i] = (_Float16)xs[i];
  if (i < (size_t)NNODES * DVZ * 3) {
    size_t n = i / 192; int rem = (int)(i % 192);
    int d = rem / 3, c = rem % 3;
    xvt16[(n * 3 + c) * 64 + d] = (_Float16)xv[i];
  }
}

// ---------------- fused GVP: modes 0=q 1=k 2=v 3=gate(scalar) 4=o(final) ----------
__global__ void __launch_bounds__(32)
k_gvp(int modeBase, const _Float16* __restrict__ WB,
      const _Float16* __restrict__ xs16, const _Float16* __restrict__ xvt16,
      const _Float16* __restrict__ outs16, const _Float16* __restrict__ outvt16,
      _Float16* __restrict__ qf16, _Float16* __restrict__ kvf16,
      _Float16* __restrict__ gates16, float* __restrict__ d_out,
      const float* qb, const float* kb, const float* vb, const float* gb,
      const float* ob, const float* qgb, const float* kgb, const float* vgb,
      const float* ogb, const float* gate_bias) {
  int mode = modeBase + blockIdx.y;
  int n0 = blockIdx.x * 16;
  int lane = threadIdx.x, hi = lane >> 4, ln = lane & 15;
  const _Float16* srcS = (mode == 4) ? outs16 : xs16;
  const _Float16* srcV = (mode == 4) ? outvt16 : xvt16;
  const _Float16* whB  = WB + (size_t)mode * 4096;
  const _Float16* wsB  = WB + 20480 + (size_t)mode * 81920;
  const _Float16* wvB  = WB + 430080 + (size_t)mode * 4096;
  const _Float16* wsvB = WB + 450560 + (size_t)mode * 16384;
  const float* wsb  = mode == 0 ? qb : mode == 1 ? kb : mode == 2 ? vb
                    : mode == 3 ? gb : ob;
  const float* wsvb = mode == 0 ? qgb : mode == 1 ? kgb : mode == 2 ? vgb : ogb;

  __shared__ __align__(16) _Float16 vhL[3][16][64];
  __shared__ __align__(16) _Float16 vnL[16][64];
  __shared__ __align__(16) _Float16 sigL[16][256];

  // ---- stage 1: vh = v^T @ wh^T  (3 x [16x64 @ 64x64]),  vn = ||vh||_c ----
#pragma unroll
  for (int nc = 0; nc < 4; nc++) {
    v8f a0 = {}, a1 = {}, a2 = {};
#pragma unroll
    for (int kc = 0; kc < 2; kc++) {
      v16h b = b_swz(whB + (kc * 4 + nc) * 512, lane);
      a0 = wmma_f16(a_frag(srcV, 64, (n0 + ln) * 3 + 0, kc * 32, hi), b, a0);
      a1 = wmma_f16(a_frag(srcV, 64, (n0 + ln) * 3 + 1, kc * 32, hi), b, a1);
      a2 = wmma_f16(a_frag(srcV, 64, (n0 + ln) * 3 + 2, kc * 32, hi), b, a2);
    }
    int col = nc * 16 + ln;
#pragma unroll
    for (int i = 0; i < 8; i++) {
      int row = i + 8 * hi;
      float s2 = a0[i] * a0[i] + a1[i] * a1[i] + a2[i] * a2[i];
      vnL[row][col] = (_Float16)sqrtf(fmaxf(s2, 1e-8f));
      vhL[0][row][col] = (_Float16)a0[i];
      vhL[1][row][col] = (_Float16)a1[i];
      vhL[2][row][col] = (_Float16)a2[i];
    }
  }
  __syncthreads();

  // ---- stage 2: s_out = [x_s | vn] @ ws^T + b   (16x320 @ 320x256) ----
  v16h aS[10];
#pragma unroll
  for (int kc = 0; kc < 8; kc++) aS[kc] = a_frag(srcS, 256, n0 + ln, kc * 32, hi);
  aS[8] = a_frag(&vnL[0][0], 64, ln, 0, hi);
  aS[9] = a_frag(&vnL[0][0], 64, ln, 32, hi);

  for (int nc = 0; nc < 16; nc++) {
    v8f acc = {};
#pragma unroll
    for (int kc = 0; kc < 10; kc++)
      acc = wmma_f16(aS[kc], b_swz(wsB + (kc * 16 + nc) * 512, lane), acc);
    int col = nc * 16 + ln;
    float bv = wsb[col];
#pragma unroll
    for (int i = 0; i < 8; i++) {
      int row = i + 8 * hi, n = n0 + row;
      float s = acc[i] + bv;
      if (mode == 3) {
        gates16[(size_t)n * 256 + col] = (_Float16)sigf(s + gate_bias[col]);
      } else {
        float sg = sigf(s);
        sigL[row][col] = (_Float16)sg;
        float sil = s * sg;
        if (mode == 4)      d_out[(size_t)n * 256 + col] = sil;
        else if (mode == 0) qf16[(size_t)n * 448 + col] = (_Float16)sil;
        else if (mode == 1) kvf16[(size_t)n * 896 + col] = (_Float16)sil;
        else                kvf16[(size_t)n * 896 + 448 + col] = (_Float16)sil;
      }
    }
  }
  if (mode == 3) return;
  __syncthreads();

  // ---- stage 3: gate = sigmoid( sigmoid(s) @ wsv^T + b )  (16x256 @ 256x64) ----
  v16h aG[8];
#pragma unroll
  for (int kc = 0; kc < 8; kc++) aG[kc] = a_frag(&sigL[0][0], 256, ln, kc * 32, hi);
  v8f gg[4];
#pragma unroll
  for (int nc = 0; nc < 4; nc++) {
    v8f acc = {};
#pragma unroll
    for (int kc = 0; kc < 8; kc++)
      acc = wmma_f16(aG[kc], b_swz(wsvB + (kc * 4 + nc) * 512, lane), acc);
    float bv = wsvb[nc * 16 + ln];
#pragma unroll
    for (int i = 0; i < 8; i++) gg[nc][i] = sigf(acc[i] + bv);
  }

  // ---- stage 4: v_out = (vh @ wv^T) * gate ----
#pragma unroll
  for (int c = 0; c < 3; c++) {
#pragma unroll
    for (int nc = 0; nc < 4; nc++) {
      v8f acc = {};
#pragma unroll
      for (int kc = 0; kc < 2; kc++)
        acc = wmma_f16(a_frag(&vhL[c][0][0], 64, ln, kc * 32, hi),
                       b_swz(wvB + (kc * 4 + nc) * 512, lane), acc);
      int dv = nc * 16 + ln;
#pragma unroll
      for (int i = 0; i < 8; i++) {
        int n = n0 + i + 8 * hi;
        float vo = acc[i] * gg[nc][i];
        if (mode == 4)      d_out[(size_t)NNODES * 256 + (size_t)n * 192 + dv * 3 + c] = vo;
        else if (mode == 0) qf16[(size_t)n * 448 + 256 + dv * 3 + c] = (_Float16)vo;
        else if (mode == 1) kvf16[(size_t)n * 896 + 256 + dv * 3 + c] = (_Float16)vo;
        else                kvf16[(size_t)n * 896 + 704 + dv * 3 + c] = (_Float16)vo;
      }
    }
  }
}

// ---------------- compression MLP + LN + softmax -> P^T [64][N] f16 --------------
__global__ void __launch_bounds__(32)
k_comp(const _Float16* __restrict__ WB, const _Float16* __restrict__ xs16,
       const float* b1, const float* lng, const float* lnb, const float* b2,
       _Float16* __restrict__ pt16) {
  const _Float16* c1B = WB + 532480;
  const _Float16* c2B = WB + 598016;
  int n0 = blockIdx.x * 16, lane = threadIdx.x, hi = lane >> 4, ln = lane & 15;
  __shared__ __align__(16) float hL[16][256];
  __shared__ __align__(16) _Float16 h2L[16][256];
  __shared__ float LL[16][64];

  v16h aC[8];
#pragma unroll
  for (int kc = 0; kc < 8; kc++) aC[kc] = a_frag(xs16, 256, n0 + ln, kc * 32, hi);
  for (int nc = 0; nc < 16; nc++) {
    v8f acc = {};
#pragma unroll
    for (int kc = 0; kc < 8; kc++)
      acc = wmma_f16(aC[kc], b_swz(c1B + (kc * 16 + nc) * 512, lane), acc);
    int col = nc * 16 + ln;
    float bv = b1[col];
#pragma unroll
    for (int i = 0; i < 8; i++) hL[i + 8 * hi][col] = acc[i] + bv;
  }
  __syncthreads();
  if (lane < 16) {
    int row = lane;
    float mu = 0.f;
    for (int j = 0; j < 256; j++) mu += hL[row][j];
    mu *= (1.0f / 256.0f);
    float var = 0.f;
    for (int j = 0; j < 256; j++) { float d = hL[row][j] - mu; var += d * d; }
    float rs = rsqrtf(var * (1.0f / 256.0f) + 1e-5f);
    for (int j = 0; j < 256; j++) {
      float xn = (hL[row][j] - mu) * rs * lng[j] + lnb[j];
      h2L[row][j] = (_Float16)(xn * sigf(xn));
    }
  }
  __syncthreads();
  v16h aH[8];
#pragma unroll
  for (int kc = 0; kc < 8; kc++) aH[kc] = a_frag(&h2L[0][0], 256, ln, kc * 32, hi);
#pragma unroll
  for (int nc = 0; nc < 4; nc++) {
    v8f acc = {};
#pragma unroll
    for (int kc = 0; kc < 8; kc++)
      acc = wmma_f16(aH[kc], b_swz(c2B + (kc * 4 + nc) * 512, lane), acc);
    int col = nc * 16 + ln;
    float bv = b2[col];
#pragma unroll
    for (int i = 0; i < 8; i++) LL[i + 8 * hi][col] = acc[i] + bv;
  }
  __syncthreads();
  if (lane < 16) {
    int row = lane, n = n0 + row;
    float mx = -1e30f;
    for (int j = 0; j < 64; j++) mx = fmaxf(mx, LL[row][j]);
    float sum = 0.f;
    for (int j = 0; j < 64; j++) sum += __expf(LL[row][j] - mx);
    float inv = 1.0f / sum;
    for (int j = 0; j < 64; j++)
      pt16[(size_t)j * NNODES + n] = (_Float16)(__expf(LL[row][j] - mx) * inv);
  }
}

// ---------------- gate_v[n] = sigmoid(mean(gate_s)) ------------------------------
__global__ void k_gatemean(const _Float16* __restrict__ g, float* __restrict__ gv) {
  int n = blockIdx.x * blockDim.x + threadIdx.x;
  if (n >= NNODES) return;
  const _Float16* p = g + (size_t)n * 256;
  float s = 0.f;
  for (int j = 0; j < 256; j++) s += (float)p[j];
  gv[n] = sigf(s * (1.0f / 256.0f));
}

__global__ void k_zero(float* p, int n) {
  int i = blockIdx.x * blockDim.x + threadIdx.x;
  if (i < n) p[i] = 0.f;
}

// ---------------- anchor reduce: ANC[b][r][f] += P^T(masked) @ Feat --------------
// Feature matrix is staged 256-node x 112-feature slabs into LDS with the Tensor
// Data Mover (TENSORcnt), then B fragments are transposed out of LDS.
__global__ void __launch_bounds__(128)
k_reduce(const _Float16* __restrict__ pt16, const _Float16* __restrict__ kvf16,
         const int* __restrict__ bi, float* __restrict__ anc) {
  int c0 = blockIdx.x * 256;
  __shared__ int biL[256];
  __shared__ __align__(16) _Float16 featL[256][112];  // 56KB slab
  for (int i = threadIdx.x; i < 256; i += 128) biL[i] = bi[c0 + i];
  __syncthreads();
  int rt = threadIdx.x >> 5, lane = threadIdx.x & 31;
  int hi = lane >> 4, ln = lane & 15;
  int bLo = biL[0], bHi = biL[255];

  for (int fg = 0; fg < 8; fg++) {  // 8 feature groups of 112
    // ---- stage slab [256 nodes x 112 halfs] into LDS ----
#if defined(__HIP_DEVICE_COMPILE__) && __has_builtin(__builtin_amdgcn_tensor_load_to_lds)
    if (threadIdx.x == 0) {
      unsigned ldsAddr = (unsigned)(size_t)(void*)&featL[0][0];
      unsigned long long ga =
          (unsigned long long)(size_t)(kvf16 + (size_t)c0 * 896 + fg * 112);
      u32x4 g0;
      g0[0] = 1u;                                   // count=1 valid descriptor
      g0[1] = ldsAddr;                              // lds_addr (bytes)
      g0[2] = (unsigned)(ga & 0xFFFFFFFFull);       // global_addr[31:0]
      g0[3] = (unsigned)((ga >> 32) & 0x1FFFFFFull) // global_addr[56:32]
              | 0x80000000u;                        // type=2 (image)
      i32x8 g1;
      g1[0] = 0x00010000;     // data_size=1 (2B elems), no multicast/flags
      g1[1] = 112 << 16;      // tensor_dim0 = 112
      g1[2] = 256 << 16;      // tensor_dim1 = 256
      g1[3] = 112 << 16;      // tile_dim0 = 112
      g1[4] = 256;            // tile_dim1 = 256 (2D tile)
      g1[5] = 896;            // tensor_dim0_stride = 896 elems
      g1[6] = 0;
      g1[7] = 0;
      i32x4 z4 = {0, 0, 0, 0};
#if __has_include(<hip/amd_detail/amd_gfx1250_TDM.h>)
      i32x8 z8 = {0, 0, 0, 0, 0, 0, 0, 0};
      __builtin_amdgcn_tensor_load_to_lds(g0, g1, z4, z4, z8, 0);
#else
      __builtin_amdgcn_tensor_load_to_lds(g0, g1, z4, z4, 0);
#endif
#if __has_builtin(__builtin_amdgcn_s_wait_tensorcnt)
      __builtin_amdgcn_s_wait_tensorcnt(0);
#else
      asm volatile("s_wait_tensorcnt 0" ::: "memory");
#endif
    }
#else
    // fallback: coalesced manual staging (14 x v8h per row)
    for (int idx = threadIdx.x; idx < 256 * 14; idx += 128) {
      int row = idx / 14, col = (idx % 14) * 8;
      *(v8h*)&featL[row][col] =
          *(const v8h*)(kvf16 + (size_t)(c0 + row) * 896 + fg * 112 + col);
    }
#endif
    __syncthreads();

    for (int bb = bLo; bb <= bHi; bb++) {
      // masked A fragments: P^T rows rt*16.., K = chunk nodes, zeroed off-batch
      v16h aR[8];
#pragma unroll
      for (int kc = 0; kc < 8; kc++) {
        const _Float16* p =
            pt16 + (size_t)(rt * 16 + ln) * NNODES + c0 + kc * 32 + 8 * hi;
        v8h lo = *(const v8h*)p;
        v8h hh = *(const v8h*)(p + 16);
        v16h r;
        int base = kc * 32 + 8 * hi;
#pragma unroll
        for (int i = 0; i < 8; i++) {
          r[i]     = (biL[base + i] == bb)      ? lo[i] : (_Float16)0.0f;
          r[8 + i] = (biL[base + 16 + i] == bb) ? hh[i] : (_Float16)0.0f;
        }
        aR[kc] = r;
      }
      for (int nc = 0; nc < 7; nc++) {
        v8f acc = {};
#pragma unroll
        for (int kc = 0; kc < 8; kc++)
          acc = wmma_f16(aR[kc], b_gather(&featL[0][0], 112, kc * 32, nc * 16, lane), acc);
        int f = fg * 112 + nc * 16 + ln;
#pragma unroll
        for (int i = 0; i < 8; i++) {
          int r = rt * 16 + i + 8 * hi;
          atomicAdd(&anc[((size_t)bb * 64 + r) * 896 + f], acc[i]);
        }
      }
    }
    __syncthreads();  // all waves done with slab before next TDM overwrite
  }
}

// ---------------- anchors fp32 -> B-frag f16 (logits side and out side) ----------
__global__ void k_anchor_fin(const float* __restrict__ anc,
                             _Float16* __restrict__ logb, _Float16* __restrict__ outb) {
  int idx = blockIdx.x * blockDim.x + threadIdx.x;
  const int per = 28672;
  if (idx >= NBATCH * per) return;
  int b = idx / per, rem = idx % per;
  int tile = rem >> 9, r2 = rem & 511;
  int lane = r2 >> 4, h = r2 & 15;
  int nn = lane & 15, kk = (h & 7) + 16 * (h >> 3) + 8 * (lane >> 4);
  {  // logitsB: K=448 (feat), N=64 (anchor): B[k=f][n=r] = ANC[b][r][f]
    int kc = tile >> 2, nc = tile & 3;
    int r = nc * 16 + nn, f = kc * 32 + kk;
    logb[idx] = (_Float16)anc[((size_t)b * 64 + r) * 896 + f];
  }
  {  // outB: K=64 (anchor), N=448 (feat): B[k=r][n=f] = ANC[b][r][448+f]
    int kc = tile / 28, nc = tile % 28;
    int r = kc * 32 + kk, f = nc * 16 + nn;
    outb[idx] = (_Float16)anc[((size_t)b * 64 + r) * 896 + 448 + f];
  }
}

// ---------------- attention + gating -> gated out_s/out_v f16 --------------------
__global__ void __launch_bounds__(32)
k_attn(const _Float16* __restrict__ qf16, const _Float16* __restrict__ logb,
       const _Float16* __restrict__ outb, const int* __restrict__ bi,
       const _Float16* __restrict__ gates16, const float* __restrict__ gatev,
       _Float16* __restrict__ outs16, _Float16* __restrict__ outvt16) {
  int n0 = blockIdx.x * 16, lane = threadIdx.x, hi = lane >> 4, ln = lane & 15;
  __shared__ float LL[16][64];
  __shared__ __align__(16) _Float16 atn[8][16][64];
  int bLo = bi[n0], bHi = bi[n0 + 15];
  int nb = bHi - bLo + 1;
  if (nb > 8) nb = 8;
  v16h aQ[14];
#pragma unroll
  for (int kc = 0; kc < 14; kc++) aQ[kc] = a_frag(qf16, 448, n0 + ln, kc * 32, hi);
  const float scale = 0.047245559f;  // 1/sqrt(448)

  for (int s = 0; s < nb; s++) {
    int bbv = bLo + s;
#pragma unroll
    for (int nc = 0; nc < 4; nc++) {
      v8f acc = {};
#pragma unroll
      for (int kc = 0; kc < 14; kc++)
        acc = wmma_f16(aQ[kc], b_swz(logb + (size_t)bbv * 28672 + (kc * 4 + nc) * 512, lane), acc);
      int col = nc * 16 + ln;
#pragma unroll
      for (int i = 0; i < 8; i++) LL[i + 8 * hi][col] = acc[i] * scale;
    }
    __syncthreads();
    if (lane < 16) {
      int row = lane;
      if (bi[n0 + row] == bbv) {
        float mx = -1e30f;
        for (int j = 0; j < 64; j++) mx = fmaxf(mx, LL[row][j]);
        float sum = 0.f;
        for (int j = 0; j < 64; j++) sum += __expf(LL[row][j] - mx);
        float inv = 1.0f / sum;
        for (int j = 0; j < 64; j++)
          atn[s][row][j] = (_Float16)(__expf(LL[row][j] - mx) * inv);
      } else {
        for (int j = 0; j < 64; j++) atn[s][row][j] = (_Float16)0.0f;
      }
    }
    __syncthreads();
  }

  for (int nc = 0; nc < 28; nc++) {
    v8f acc = {};
    for (int s = 0; s < nb; s++) {
#pragma unroll
      for (int kc = 0; kc < 2; kc++) {
        v16h a = a_frag(&atn[s][0][0], 64, ln, kc * 32, hi);
        acc = wmma_f16(a, b_swz(outb + (size_t)(bLo + s) * 28672 + (kc * 28 + nc) * 512, lane), acc);
      }
    }
    int f = nc * 16 + ln;
#pragma unroll
    for (int i = 0; i < 8; i++) {
      int n = n0 + i + 8 * hi;
      if (nc < 16) {
        float g = (float)gates16[(size_t)n * 256 + f];
        outs16[(size_t)n * 256 + f] = (_Float16)(acc[i] * g);
      } else {
        int idx = f - 256, dv = idx / 3, c = idx % 3;
        outvt16[((size_t)n * 3 + c) * 64 + dv] = (_Float16)(acc[i] * gatev[n]);
      }
    }
  }
}

// =================================================================================
extern "C" void kernel_launch(void* const* d_in, const int* in_sizes, int n_in,
                              void* d_out, int out_size, void* d_ws, size_t ws_size,
                              hipStream_t stream) {
  (void)in_sizes; (void)n_in; (void)out_size; (void)ws_size;
  const float* x_s = (const float*)d_in[0];
  const float* x_v = (const float*)d_in[1];
  const int* bidx  = (const int*)d_in[2];

  // workspace layout (half units)
  _Float16* wsH = (_Float16*)d_ws;
  size_t off = 0;
  _Float16* WB      = wsH + off; off += 614400;
  _Float16* XS16    = wsH + off; off += (size_t)NNODES * 256;
  _Float16* XVT16   = wsH + off; off += (size_t)NNODES * 192;
  _Float16* QF16    = wsH + off; off += (size_t)NNODES * 448;
  _Float16* KVF16   = wsH + off; off += (size_t)NNODES * 896;
  _Float16* PT16    = wsH + off; off += (size_t)64 * NNODES;
  _Float16* GATES16 = wsH + off; off += (size_t)NNODES * 256;
  _Float16* OUTS16  = wsH + off; off += (size_t)NNODES * 256;
  _Float16* OUTVT16 = wsH + off; off += (size_t)NNODES * 192;
  _Float16* LOGB16  = wsH + off; off += (size_t)NBATCH * 28672;
  _Float16* OUTB16  = wsH + off; off += (size_t)NBATCH * 28672;
  float* GATEV  = (float*)(wsH + off); off += 2 * (size_t)NNODES;
  float* ANCF32 = (float*)(wsH + off); off += 2 * (size_t)NBATCH * 64 * 896;

  auto swz = [&](int inIdx, _Float16* dst, int K, int Nout) {
    int total = K * Nout;
    k_swz<<<(total + 255) / 256, 256, 0, stream>>>((const float*)d_in[inIdx], dst, K, Nout);
  };
  // slots: 0=q 1=k 2=v 3=g 4=o
  const int whIn[5]  = {3, 9, 15, 33, 21};
  const int wsIn[5]  = {4, 10, 16, 34, 22};
  const int wvIn[5]  = {6, 12, 18, -1, 24};
  const int wsvIn[5] = {7, 13, 19, -1, 25};
  for (int m = 0; m < 5; m++) {
    swz(whIn[m], WB + (size_t)m * 4096, 64, 64);
    swz(wsIn[m], WB + 20480 + (size_t)m * 81920, 320, 256);
    if (wvIn[m] >= 0)  swz(wvIn[m],  WB + 430080 + (size_t)m * 4096, 64, 64);
    if (wsvIn[m] >= 0) swz(wsvIn[m], WB + 450560 + (size_t)m * 16384, 256, 64);
  }
  swz(27, WB + 532480, 256, 256);  // comp_w1
  swz(31, WB + 598016, 256, 64);   // comp_w2

  {  // input conversion
    size_t total = (size_t)NNODES * 256;
    k_prep_x<<<(unsigned)((total + 255) / 256), 256, 0, stream>>>(x_s, x_v, XS16, XVT16);
  }

  const float* qb  = (const float*)d_in[5];
  const float* kb  = (const float*)d_in[11];
  const float* vb  = (const float*)d_in[17];
  const float* gb  = (const float*)d_in[35];
  const float* ob  = (const float*)d_in[23];
  const float* qgb = (const float*)d_in[8];
  const float* kgb = (const float*)d_in[14];
  const float* vgb = (const float*)d_in[20];
  const float* ogb = (const float*)d_in[26];
  const float* gbias = (const float*)d_in[36];

  // Q/K/V + gate GVPs (4096 tiles x 4 projections, 1 wave each)
  k_gvp<<<dim3(NNODES / 16, 4), 32, 0, stream>>>(
      0, WB, XS16, XVT16, OUTS16, OUTVT16, QF16, KVF16, GATES16, (float*)d_out,
      qb, kb, vb, gb, ob, qgb, kgb, vgb, ogb, gbias);

  k_gatemean<<<(NNODES + 255) / 256, 256, 0, stream>>>(GATES16, GATEV);

  k_comp<<<NNODES / 16, 32, 0, stream>>>(
      WB, XS16, (const float*)d_in[28], (const float*)d_in[29],
      (const float*)d_in[30], (const float*)d_in[32], PT16);

  k_zero<<<(NBATCH * 64 * 896 + 255) / 256, 256, 0, stream>>>(ANCF32, NBATCH * 64 * 896);

  k_reduce<<<NNODES / 256, 128, 0, stream>>>(PT16, KVF16, bidx, ANCF32);

  k_anchor_fin<<<(NBATCH * 28672 + 255) / 256, 256, 0, stream>>>(ANCF32, LOGB16, OUTB16);

  k_attn<<<NNODES / 16, 32, 0, stream>>>(QF16, LOGB16, OUTB16, bidx, GATES16, GATEV,
                                         OUTS16, OUTVT16);

  // output GVP (mode 4) -> final d_out (silu(s), gated v)
  k_gvp<<<dim3(NNODES / 16, 1), 32, 0, stream>>>(
      4, WB, XS16, XVT16, OUTS16, OUTVT16, QF16, KVF16, GATES16, (float*)d_out,
      qb, kb, vb, gb, ob, qgb, kgb, vgb, ogb, gbias);
}
